// QaMultiSrcPointerGen_63161789055223
// MI455X (gfx1250) — matrix operand order repaired and checked by reading, not compiled, not run on hardware
//
#include <hip/hip_runtime.h>
#include <hip/hip_bf16.h>

// ---------------- problem constants (from reference) ----------------
#define NB   8
#define TT   80
#define KL   1000
#define JJ   50
#define KK   5
#define DD   256
#define DE   300
#define DV   30000
#define DEXT 35000
#define LP   (KL / KK)          // 200
#define PQ   (KL + JJ)          // 1050
#define ROWS (NB * TT)          // 640
#define OUT_EXT_ELEMS ((long long)ROWS * DEXT)   // 22,400,000
#define OUT_TOTAL     (OUT_EXT_ELEMS + (long long)ROWS * 3)

typedef __attribute__((ext_vector_type(16))) __bf16 v16bf;
typedef __attribute__((ext_vector_type(8)))  float  v8f;

// Pack two f32 -> two bf16 (RNE) in one dword.
#if __has_builtin(__builtin_amdgcn_cvt_pk_bf16_f32)
typedef __attribute__((ext_vector_type(2))) __bf16 v2bf;
static __device__ __forceinline__ unsigned pack2bf(float a, float b) {
  v2bf p = __builtin_amdgcn_cvt_pk_bf16_f32(a, b);   // v_cvt_pk_bf16_f32
  return __builtin_bit_cast(unsigned, p);
}
#else
static __device__ __forceinline__ unsigned pack2bf(float a, float b) {
  unsigned u0 = __builtin_bit_cast(unsigned, a);
  unsigned u1 = __builtin_bit_cast(unsigned, b);
  u0 += 0x7FFFu + ((u0 >> 16) & 1u);                 // RNE
  u1 += 0x7FFFu + ((u1 >> 16) & 1u);
  return __builtin_amdgcn_perm(u1, u0, 0x07060302u); // v_perm_b32: [u0.hi, u1.hi]
}
#endif

union Frag16 {            // one WMMA 16-bf16 fragment as 8 packed dwords
  v16bf v;
  unsigned u[8];
};

static __device__ __forceinline__ float waveReduceSum(float v) {
  #pragma unroll
  for (int o = 16; o > 0; o >>= 1) v += __shfl_down(v, o, 32);
  return v;
}

// ---------------------------------------------------------------
// Generic batched GEMM: C[M x Nc] = A[M x Kd] * B[Kd x Nc] (+bias)
// bf16 WMMA, fp32 accumulate. One wave = one 16x64 output strip
// (4 accumulators share one A fragment -> 4 back-to-back v_wmma).
// LDB is a template constant so all B loads are immediate-offset
// loads from one stable base (no per-load address chains / xcnt).
// ---------------------------------------------------------------
template <int LDB>
__global__ __launch_bounds__(128)
void wmma_gemm_bf16(const float* __restrict__ A, long long lda, long long strideA,
                    const float* __restrict__ B, long long strideB,
                    const float* __restrict__ bias,
                    float* __restrict__ C, long long ldc, long long strideC,
                    int M, int Nc, int Kd) {
  const int batch = blockIdx.y;
  A += (long long)batch * strideA;
  B += (long long)batch * strideB;
  C += (long long)batch * strideC;

  const int tilesN = (Nc + 63) >> 6;              // 64-wide strips
  const int tilesM = (M + 15) >> 4;
  const int tile = blockIdx.x * (blockDim.x >> 5) + (threadIdx.x >> 5);
  if (tile >= tilesM * tilesN) return;            // wave-uniform exit

  const int tm = tile / tilesN;
  const int tn = tile - tm * tilesN;
  const int lane = threadIdx.x & 31;
  const int half = lane >> 4;                     // lane half (0/1)
  const int l15  = lane & 15;
  const int row0 = tm << 4;
  const int col0 = tn << 6;

  const int kbA = half ? 16 : 0;                  // A-fragment K base (contiguous k = 16*half + s)
  const int kbB = half ? 16 : 0;                  // B-fragment K base (same contiguous k = 16*half + s)

  const int arow = min(row0 + l15, M - 1);        // clamp: dead rows masked at store
  const float* Ak = A + (long long)arow * lda + kbA;   // bumped by 32 per k-step

  int ccol[4];
  const float* Bk[4];                             // bumped by 32*LDB per k-step
  #pragma unroll
  for (int s = 0; s < 4; ++s) {
    ccol[s] = col0 + 16 * s + l15;
    Bk[s] = B + min(ccol[s], Nc - 1) + (long long)kbB * LDB;  // clamp dead cols
  }

  v8f acc[4] = {};
  if (bias) {
    #pragma unroll
    for (int s = 0; s < 4; ++s) {
      const float bv = bias[min(ccol[s], Nc - 1)];
      #pragma unroll
      for (int g = 0; g < 8; ++g) acc[s][g] = bv;
    }
  }

  const int Kfull = Kd & ~31;
  for (int k0 = 0; k0 < Kfull; k0 += 32) {
    // ---- A fragment: two runs of 8 consecutive K, float2 loads, const offsets ----
    Frag16 af;
    #pragma unroll
    for (int g = 0; g < 4; ++g) {
      const float2 x = *(const float2*)(Ak + 2 * g);          // offsets 0..7
      af.u[g] = pack2bf(x.x, x.y);
    }
    #pragma unroll
    for (int g = 0; g < 4; ++g) {
      const float2 x = *(const float2*)(Ak + 8 + 2 * g);      // offsets 8..15
      af.u[4 + g] = pack2bf(x.x, x.y);
    }
    Ak += 32;

    // speculative prefetch of next B block (dropped if OOB)
    __builtin_prefetch(Bk[0] + (long long)32 * LDB, 0, 1);

    #pragma unroll
    for (int s = 0; s < 4; ++s) {
      Frag16 bf;
      #pragma unroll
      for (int g = 0; g < 8; ++g) {
        // compile-time byte offsets: (2g)*LDB*4 and (2g+1)*LDB*4
        const float b0 = Bk[s][(2 * g) * LDB];
        const float b1 = Bk[s][(2 * g + 1) * LDB];
        bf.u[g] = pack2bf(b0, b1);
      }
      acc[s] = __builtin_amdgcn_wmma_f32_16x16x32_bf16(false, af.v, false, bf.v,
                                                       (short)0, acc[s], false, false);
    }
    #pragma unroll
    for (int s = 0; s < 4; ++s) Bk[s] += (long long)32 * LDB;
  }

  if (Kfull < Kd) {                               // single guarded K tail
    const int k0 = Kfull;
    const float* Arow = A + (long long)arow * lda;
    Frag16 af;
    #pragma unroll
    for (int g = 0; g < 8; ++g) {
      const int kk = k0 + kbA + 2 * g;
      float f0 = Arow[min(kk, Kd - 1)];     f0 = (kk     < Kd) ? f0 : 0.f;
      float f1 = Arow[min(kk + 1, Kd - 1)]; f1 = (kk + 1 < Kd) ? f1 : 0.f;
      af.u[g] = pack2bf(f0, f1);
    }
    #pragma unroll
    for (int s = 0; s < 4; ++s) {
      const float* Bc = B + min(ccol[s], Nc - 1);
      Frag16 bf;
      #pragma unroll
      for (int g = 0; g < 8; ++g) {
        const int kk = k0 + kbB + 2 * g;
        float f0 = Bc[(long long)min(kk, Kd - 1) * LDB];     f0 = (kk     < Kd) ? f0 : 0.f;
        float f1 = Bc[(long long)min(kk + 1, Kd - 1) * LDB]; f1 = (kk + 1 < Kd) ? f1 : 0.f;
        bf.u[g] = pack2bf(f0, f1);
      }
      acc[s] = __builtin_amdgcn_wmma_f32_16x16x32_bf16(false, af.v, false, bf.v,
                                                       (short)0, acc[s], false, false);
    }
  }

  #pragma unroll
  for (int s = 0; s < 4; ++s) {
    if (ccol[s] < Nc) {
      #pragma unroll
      for (int g = 0; g < 8; ++g) {
        const int r = row0 + g + half * 8;        // C/D layout: vgpr g + 8*half
        if (r < M) C[(long long)r * ldc + ccol[s]] = acc[s][g];
      }
    }
  }
}

// ---------------------------------------------------------------
// Additive attention scores vs Mq:  a_q[b,t,j] = softmax_j( v . tanh(hq+hk) + vb )
// one block (256 thr, 8 waves) per (b,t) row
// ---------------------------------------------------------------
__global__ void scores_q_kernel(const float* __restrict__ hq,   // (ROWS, D)
                                const float* __restrict__ hk,   // (N*J, D)
                                const float* __restrict__ v, const float* __restrict__ vb,
                                const unsigned char* __restrict__ mask,  // (N*J)
                                float* __restrict__ a_q) {      // (ROWS, J)
  const int r = blockIdx.x;              // b*T + t
  const int b = r / TT;
  const int tid = threadIdx.x, lane = tid & 31, wave = tid >> 5;
  __shared__ float sh_hq[DD];
  __shared__ float sh_e[JJ];
  __shared__ float red[256];

  sh_hq[tid] = hq[(long long)r * DD + tid];
  __syncthreads();

  for (int j = wave; j < JJ; j += 8) {
    const float* hkr = hk + (long long)(b * JJ + j) * DD;
    float s = 0.f;
    for (int d = lane; d < DD; d += 32) s += tanhf(sh_hq[d] + hkr[d]) * v[d];
    s = waveReduceSum(s);
    if (lane == 0) {
      s += vb[0];
      if (!mask[b * JJ + j]) s = -1e9f;
      sh_e[j] = s;
    }
  }
  __syncthreads();

  float m = -1e30f;
  for (int j = tid; j < JJ; j += 256) m = fmaxf(m, sh_e[j]);
  red[tid] = m; __syncthreads();
  for (int o = 128; o > 0; o >>= 1) { if (tid < o) red[tid] = fmaxf(red[tid], red[tid + o]); __syncthreads(); }
  const float gmax = red[0]; __syncthreads();

  float s = 0.f;
  for (int j = tid; j < JJ; j += 256) { const float e = expf(sh_e[j] - gmax); sh_e[j] = e; s += e; }
  red[tid] = s; __syncthreads();
  for (int o = 128; o > 0; o >>= 1) { if (tid < o) red[tid] += red[tid + o]; __syncthreads(); }
  const float inv = 1.f / red[0];
  for (int j = tid; j < JJ; j += 256) a_q[(long long)r * JJ + j] = sh_e[j] * inv;
}

// ---------------------------------------------------------------
// Additive attention scores vs Mp with beta modulation + renorm
// mod_a_p = beta_l * exp(e_l - max) / sum(beta * exp)
// ---------------------------------------------------------------
__global__ void scores_p_kernel(const float* __restrict__ hq,   // (ROWS, D)
                                const float* __restrict__ hk,   // (N*KL, D)
                                const float* __restrict__ v, const float* __restrict__ vb,
                                const unsigned char* __restrict__ mask,  // (N*KL)
                                const float* __restrict__ betas,         // (N, K)
                                float* __restrict__ mod_a_p) {  // (ROWS, KL)
  const int r = blockIdx.x;
  const int b = r / TT;
  const int tid = threadIdx.x, lane = tid & 31, wave = tid >> 5;
  __shared__ float sh_hq[DD];
  __shared__ float sh_e[KL];
  __shared__ float red[256];

  sh_hq[tid] = hq[(long long)r * DD + tid];
  __syncthreads();

  for (int l = wave; l < KL; l += 8) {
    const float* hkr = hk + (long long)(b * KL + l) * DD;
    float s = 0.f;
    for (int d = lane; d < DD; d += 32) s += tanhf(sh_hq[d] + hkr[d]) * v[d];
    s = waveReduceSum(s);
    if (lane == 0) {
      s += vb[0];
      if (!mask[b * KL + l]) s = -1e9f;
      sh_e[l] = s;
    }
  }
  __syncthreads();

  float m = -1e30f;
  for (int l = tid; l < KL; l += 256) m = fmaxf(m, sh_e[l]);
  red[tid] = m; __syncthreads();
  for (int o = 128; o > 0; o >>= 1) { if (tid < o) red[tid] = fmaxf(red[tid], red[tid + o]); __syncthreads(); }
  const float gmax = red[0]; __syncthreads();

  float s = 0.f;
  for (int l = tid; l < KL; l += 256) {
    const float w = betas[b * KK + l / LP] * expf(sh_e[l] - gmax);
    sh_e[l] = w; s += w;
  }
  red[tid] = s; __syncthreads();
  for (int o = 128; o > 0; o >>= 1) { if (tid < o) red[tid] += red[tid + o]; __syncthreads(); }
  const float inv = 1.f / red[0];
  for (int l = tid; l < KL; l += 256) mod_a_p[(long long)r * KL + l] = sh_e[l] * inv;
}

// ---------------------------------------------------------------
// Mixture lambdas: softmax3([Ma,ctx_q,ctx_p] @ mix_W + mix_b)
// 96 threads (3 waves) per row; wave w owns component w
// ---------------------------------------------------------------
__global__ void mix_kernel(const float* __restrict__ Ma,
                           const float* __restrict__ ctx_q,
                           const float* __restrict__ ctx_p,
                           const float* __restrict__ mix_W,   // (768,3)
                           const float* __restrict__ mix_b,
                           float* __restrict__ lam_ws,        // (ROWS,3)
                           float* __restrict__ lam_out) {     // (ROWS,3)
  const int r = blockIdx.x;
  const int lane = threadIdx.x & 31, comp = threadIdx.x >> 5;
  __shared__ float e[3];
  float s = 0.f;
  for (int d = lane; d < 3 * DD; d += 32) {
    float x;
    if (d < DD)            x = Ma[(long long)r * DD + d];
    else if (d < 2 * DD)   x = ctx_q[(long long)r * DD + (d - DD)];
    else                   x = ctx_p[(long long)r * DD + (d - 2 * DD)];
    s += x * mix_W[d * 3 + comp];
  }
  s = waveReduceSum(s);
  if (lane == 0) e[comp] = s + mix_b[comp];
  __syncthreads();
  if (threadIdx.x == 0) {
    const float m = fmaxf(e[0], fmaxf(e[1], e[2]));
    const float e0 = expf(e[0] - m), e1 = expf(e[1] - m), e2 = expf(e[2] - m);
    const float inv = 1.f / (e0 + e1 + e2);
    lam_ws[r * 3 + 0] = e0 * inv; lam_ws[r * 3 + 1] = e1 * inv; lam_ws[r * 3 + 2] = e2 * inv;
    lam_out[r * 3 + 0] = e0 * inv; lam_out[r * 3 + 1] = e1 * inv; lam_out[r * 3 + 2] = e2 * inv;
  }
}

// ---------------------------------------------------------------
// In-place vocab softmax: out rows hold raw logits in cols [0,DV);
// apply SPECIAL_IDX=-10, softmax over DV, scale by lambda_v.
// ---------------------------------------------------------------
__global__ void vocab_softmax_kernel(float* __restrict__ out,
                                     const float* __restrict__ lam) {
  const int r = blockIdx.x;
  float* row = out + (long long)r * DEXT;
  const float lamv = lam[r * 3 + 0];
  const int tid = threadIdx.x;
  __shared__ float red[256];

  float m = -1e30f;
  for (int c = tid; c < DV; c += 256) {
    const float v = (c < 4) ? -10.f : row[c];
    m = fmaxf(m, v);
  }
  red[tid] = m; __syncthreads();
  for (int o = 128; o > 0; o >>= 1) { if (tid < o) red[tid] = fmaxf(red[tid], red[tid + o]); __syncthreads(); }
  const float gmax = red[0]; __syncthreads();

  float s = 0.f;
  for (int c = tid; c < DV; c += 256) {
    const float v = (c < 4) ? -10.f : row[c];
    s += expf(v - gmax);
  }
  red[tid] = s; __syncthreads();
  for (int o = 128; o > 0; o >>= 1) { if (tid < o) red[tid] += red[tid + o]; __syncthreads(); }
  const float scale = lamv / red[0];

  for (int c = tid; c < DV; c += 256) {
    const float v = (c < 4) ? -10.f : row[c];
    row[c] = expf(v - gmax) * scale;
  }
}

// ---------------------------------------------------------------
// Scatter-add pointer probs into extended vocab
// ---------------------------------------------------------------
__global__ void scatter_kernel(const float* __restrict__ mod_a_p,
                               const float* __restrict__ a_q,
                               const float* __restrict__ lam,
                               const int* __restrict__ pq_ext,   // (N, PQ)
                               float* __restrict__ out) {
  const long long idx = (long long)blockIdx.x * blockDim.x + threadIdx.x;
  if (idx >= (long long)NB * TT * PQ) return;
  const int b = (int)(idx / ((long long)TT * PQ));
  const int rem = (int)(idx - (long long)b * TT * PQ);
  const int t = rem / PQ;
  const int i = rem - t * PQ;
  const int r = b * TT + t;
  float val;
  if (i < KL) val = mod_a_p[(long long)r * KL + i] * lam[r * 3 + 2];    // lambda_p
  else        val = a_q[(long long)r * JJ + (i - KL)] * lam[r * 3 + 1]; // lambda_q
  const int col = pq_ext[b * PQ + i];
  atomicAdd(out + (long long)r * DEXT + col, val);
}

__global__ void zero_kernel(float4* __restrict__ p, long long n4) {
  const float4 z = {0.f, 0.f, 0.f, 0.f};
  for (long long i = (long long)blockIdx.x * blockDim.x + threadIdx.x;
       i < n4; i += (long long)gridDim.x * blockDim.x)
    p[i] = z;
}

// ---------------------------------------------------------------
extern "C" void kernel_launch(void* const* d_in, const int* in_sizes, int n_in,
                              void* d_out, int out_size, void* d_ws, size_t ws_size,
                              hipStream_t stream) {
  const float* Mp    = (const float*)d_in[0];
  const float* Mq    = (const float*)d_in[1];
  const float* Ma    = (const float*)d_in[2];
  const unsigned char* mask_p = (const unsigned char*)d_in[3];
  const unsigned char* mask_q = (const unsigned char*)d_in[4];
  const float* betas = (const float*)d_in[5];
  const int*   pq_ext = (const int*)d_in[6];
  const float* q_Wk = (const float*)d_in[8];
  const float* q_bk = (const float*)d_in[9];
  const float* q_Wq = (const float*)d_in[10];
  const float* q_bq = (const float*)d_in[11];
  const float* q_v  = (const float*)d_in[12];
  const float* q_vb = (const float*)d_in[13];
  const float* p_Wk = (const float*)d_in[14];
  const float* p_bk = (const float*)d_in[15];
  const float* p_Wq = (const float*)d_in[16];
  const float* p_bq = (const float*)d_in[17];
  const float* p_v  = (const float*)d_in[18];
  const float* p_vb = (const float*)d_in[19];
  const float* vg_W1 = (const float*)d_in[20];
  const float* vg_b1 = (const float*)d_in[21];
  const float* vg_W2 = (const float*)d_in[22];
  const float* mix_W = (const float*)d_in[23];
  const float* mix_b = (const float*)d_in[24];

  float* out = (float*)d_out;

  // workspace layout (floats), ~14.7 MB total
  float* w = (float*)d_ws;
  float* hkq     = w;                       // 400*256
  float* hqq     = hkq  + 400 * DD;         // 640*256
  float* hkp     = hqq  + ROWS * DD;        // 8000*256
  float* hqp     = hkp  + NB * KL * DD;     // 640*256
  float* hidden  = hqp  + ROWS * DD;        // 640*300
  float* a_q     = hidden + ROWS * DE;      // 640*50
  float* mod_a_p = a_q  + ROWS * JJ;        // 640*1000
  float* ctx_q   = mod_a_p + ROWS * KL;     // 640*256
  float* ctx_p   = ctx_q + ROWS * DD;       // 640*256
  float* lam_ws  = ctx_p + ROWS * DD;       // 640*3
  (void)in_sizes; (void)n_in; (void)out_size; (void)ws_size;

  auto gemm = [&](const float* A, long long lda, long long sA,
                  const float* B, int ldb, long long sB,
                  const float* bias,
                  float* C, long long ldc, long long sC,
                  int M, int Nc, int Kd, int batches) {
    const int tiles = ((M + 15) / 16) * ((Nc + 63) / 64);
    dim3 grid((tiles + 3) / 4, batches);
    if (ldb == DD)
      wmma_gemm_bf16<DD><<<grid, 128, 0, stream>>>(A, lda, sA, B, sB, bias, C, ldc, sC, M, Nc, Kd);
    else if (ldb == DE)
      wmma_gemm_bf16<DE><<<grid, 128, 0, stream>>>(A, lda, sA, B, sB, bias, C, ldc, sC, M, Nc, Kd);
    else
      wmma_gemm_bf16<DV><<<grid, 128, 0, stream>>>(A, lda, sA, B, sB, bias, C, ldc, sC, M, Nc, Kd);
  };

  // 0) zero whole output (ext region + lambda tail); OUT_TOTAL % 4 == 0
  zero_kernel<<<4096, 256, 0, stream>>>((float4*)out, OUT_TOTAL / 4);

  // 1) projections (bf16 WMMA, shared weights -> single batch over N*rows)
  gemm(Mq, DD, 0, q_Wk, DD, 0, q_bk, hkq, DD, 0, NB * JJ, DD, DD, 1);
  gemm(Ma, DD, 0, q_Wq, DD, 0, q_bq, hqq, DD, 0, ROWS,    DD, DD, 1);
  gemm(Mp, DD, 0, p_Wk, DD, 0, p_bk, hkp, DD, 0, NB * KL, DD, DD, 1);
  gemm(Ma, DD, 0, p_Wq, DD, 0, p_bq, hqp, DD, 0, ROWS,    DD, DD, 1);
  gemm(Ma, DD, 0, vg_W1, DE, 0, vg_b1, hidden, DE, 0, ROWS, DE, DD, 1);

  // 2) additive attention scores (tanh / TRANS-unit heavy)
  scores_q_kernel<<<ROWS, 256, 0, stream>>>(hqq, hkq, q_v, q_vb, mask_q, a_q);
  scores_p_kernel<<<ROWS, 256, 0, stream>>>(hqp, hkp, p_v, p_vb, mask_p, betas, mod_a_p);

  // 3) contexts (batched WMMA GEMM: per-batch A/B/C pointers via strides)
  gemm(a_q,     JJ, (long long)TT * JJ, Mq, DD, (long long)JJ * DD, nullptr,
       ctx_q, DD, (long long)TT * DD, TT, DD, JJ, NB);
  gemm(mod_a_p, KL, (long long)TT * KL, Mp, DD, (long long)KL * DD, nullptr,
       ctx_p, DD, (long long)TT * DD, TT, DD, KL, NB);

  // 4) mixture lambdas (also writes output tail)
  mix_kernel<<<ROWS, 96, 0, stream>>>(Ma, ctx_q, ctx_p, mix_W, mix_b,
                                      lam_ws, out + OUT_EXT_ELEMS);

  // 5) vocab logits GEMM straight into out[:, 0:DV] (ldc = DEXT), then
  //    in-place softmax * lambda_v, then scatter-add pointer probs.
  gemm(hidden, DE, 0, vg_W2, DV, 0, nullptr, out, DEXT, 0, ROWS, DV, DE, 1);
  vocab_softmax_kernel<<<ROWS, 256, 0, stream>>>(out, lam_ws);

  const long long sc_threads = (long long)NB * TT * PQ;
  scatter_kernel<<<(int)((sc_threads + 255) / 256), 256, 0, stream>>>(
      mod_a_p, a_q, lam_ws, pq_ext, out);
}
